// Net_13486197310235
// MI455X (gfx1250) — compile-verified, matrix operand order checked by reading, not run on hardware
//
#include <hip/hip_runtime.h>

// ---------------------------------------------------------------------------
// Types
// ---------------------------------------------------------------------------
typedef __attribute__((ext_vector_type(16))) __bf16 v16bf;
typedef __attribute__((ext_vector_type(2)))  __bf16 v2bf;
typedef __attribute__((ext_vector_type(8)))  float  v8f;
typedef __attribute__((ext_vector_type(4)))  unsigned int u32x4;
typedef int vsi4 __attribute__((vector_size(16)));   // matches async-LDS builtin param

union FragU { v16bf v; u32x4 u[2]; };

#if __has_builtin(__builtin_amdgcn_global_load_async_to_lds_b128)
#define HAVE_ASYNC_LDS 1
#endif

#define AS1 __attribute__((address_space(1)))
#define AS3 __attribute__((address_space(3)))

__device__ __forceinline__ unsigned short f2bf(float f) {
    unsigned u = __float_as_uint(f);
    unsigned r = (u + 0x7FFFu + ((u >> 16) & 1u)) >> 16;
    return (unsigned short)r;
}

// pack two f32 -> packed bf16 pair (RNE)
__device__ __forceinline__ unsigned pkbf(float lo, float hi) {
#if __has_builtin(__builtin_amdgcn_cvt_pk_bf16_f32)
    union { v2bf v; unsigned u; } c;
    c.v = __builtin_amdgcn_cvt_pk_bf16_f32(lo, hi);
    return c.u;
#else
    return (unsigned)f2bf(lo) | ((unsigned)f2bf(hi) << 16);
#endif
}

__device__ __forceinline__ void wait_async_all() {
#ifdef HAVE_ASYNC_LDS
#if __has_builtin(__builtin_amdgcn_s_wait_asynccnt)
    __builtin_amdgcn_s_wait_asynccnt(0);
#else
    asm volatile("s_wait_asynccnt 0x0" ::: "memory");
#endif
#endif
}

// ---------------------------------------------------------------------------
// Generic bf16 WMMA GEMM with up-to-3-segment gathered A input.
//   out[M, Nout] = act( concat_A[M, Ktot] @ W[Ktot, Nout] + bias )
// Segment widths must be multiples of 16 (host guarantees this; the global-
// model concat is zero-padded to 160). Weights pre-converted to bf16,
// transposed+zero-padded: wt[Nout, Kpad], Kpad = ceil32(Ktot).
// Block: 256 threads (8 waves). Block tile 128x128, K-step 32, double-buffered
// LDS. A tiles: global->reg(f32)->cvt->LDS pipeline. B tiles: async direct
// global->LDS (ASYNCcnt) when the toolchain exposes the CDNA5 builtin.
// Wave (wr, wc): rows [wr*32,+32), cols [wc*64,+64) -> 2x4 wmma tiles.
// ---------------------------------------------------------------------------
#define TSTR 40  // LDS row stride in shorts (80B, 16B aligned)

__global__ __launch_bounds__(256) void gemm_bf16_seg(
    const float* __restrict__ a0, int w0, const int* __restrict__ i0,
    const float* __restrict__ a1, int w1, const int* __restrict__ i1,
    const float* __restrict__ a2, int w2, const int* __restrict__ i2,
    const unsigned short* __restrict__ wt, const float* __restrict__ bias,
    float* __restrict__ out, int M, int Ktot, int Nout, int relu)
{
    __shared__ __align__(16) unsigned short lA[2][128 * TSTR];
    __shared__ __align__(16) unsigned short lB[2][128 * TSTR];

    const int tid  = threadIdx.x;
    const int lane = tid & 31;
    const int wave = tid >> 5;
    const int wr   = wave & 3;
    const int wc   = wave >> 2;
    const long m0  = (long)blockIdx.y * 128;
    const int  n0  = blockIdx.x * 128;
    const int Kpad = (Ktot + 31) & ~31;
    const int w01  = w0 + w1;

    // Per-thread A/B-tile assignment: row rl (=col for B), 16-wide K chunk ch
    const int rl = tid >> 1;
    const int ch = (tid & 1) * 16;
    const long rg = m0 + rl;
    const bool rv = rg < (long)M;
    long r0 = 0, r1 = 0, r2 = 0;
    if (rv) {
        r0 = i0 ? (long)i0[rg] : rg;
        if (w1) r1 = i1 ? (long)i1[rg] : rg;
        if (w2) r2 = i2 ? (long)i2[rg] : rg;
    }
    const unsigned short* __restrict__ wrow = wt + (size_t)(n0 + rl) * Kpad + ch;

    // Pipeline registers (raw f32 for A so conversion happens at store time,
    // i.e. after the wmma issue for the current tile)
    float fa[16];
#ifndef HAVE_ASYNC_LDS
    u32x4 fb0, fb1;
#endif

    auto fetchA = [&](int kt) {
        const int kb = kt + ch;                 // chunk base
        const int rem = Ktot - kb;              // >=16 -> full fast chunk
        if (rv && rem >= 16) {
            // chunk fully inside one segment (widths are multiples of 16)
            const float* __restrict__ p;
            if (kb < w0)       p = a0 + r0 * (long)w0 + kb;
            else if (kb < w01) p = a1 + r1 * (long)w1 + (kb - w0);
            else               p = a2 + r2 * (long)w2 + (kb - w01);
            const float4* v = (const float4*)p;
            float4 q0 = v[0], q1 = v[1], q2 = v[2], q3 = v[3];
            fa[0] = q0.x;  fa[1] = q0.y;  fa[2] = q0.z;  fa[3] = q0.w;
            fa[4] = q1.x;  fa[5] = q1.y;  fa[6] = q1.z;  fa[7] = q1.w;
            fa[8] = q2.x;  fa[9] = q2.y;  fa[10] = q2.z; fa[11] = q2.w;
            fa[12] = q3.x; fa[13] = q3.y; fa[14] = q3.z; fa[15] = q3.w;
        } else {
            // tail / out-of-range chunk (rare): per-element with zero fill
            #pragma unroll
            for (int j = 0; j < 16; j++) {
                const int k = kb + j;
                float v = 0.0f;
                if (rv && k < Ktot) {
                    if (k < w0)       v = a0[r0 * (long)w0 + k];
                    else if (k < w01) v = a1[r1 * (long)w1 + (k - w0)];
                    else              v = a2[r2 * (long)w2 + (k - w01)];
                }
                fa[j] = v;
            }
        }
    };

    // B tile: either async direct global->LDS, or staged through registers
    auto issueB = [&](int kt, int buf) {
        const unsigned short* src = wrow + kt;
        unsigned short* dst = &lB[buf][rl * TSTR + ch];
#ifdef HAVE_ASYNC_LDS
        __builtin_amdgcn_global_load_async_to_lds_b128(
            (AS1 vsi4*)(unsigned long long)(size_t)src,
            (AS3 vsi4*)dst, 0, 0);
        __builtin_amdgcn_global_load_async_to_lds_b128(
            (AS1 vsi4*)(unsigned long long)(size_t)(src + 8),
            (AS3 vsi4*)(dst + 8), 0, 0);
#else
        fb0 = *(const u32x4*)(src);
        fb1 = *(const u32x4*)(src + 8);
        (void)dst;
#endif
    };

    auto storeTile = [&](int buf) {
        u32x4 p0, p1;
        #pragma unroll
        for (int i = 0; i < 4; i++)
            p0[i] = pkbf(fa[2 * i], fa[2 * i + 1]);
        #pragma unroll
        for (int i = 0; i < 4; i++)
            p1[i] = pkbf(fa[8 + 2 * i], fa[9 + 2 * i]);
        *(u32x4*)&lA[buf][rl * TSTR + ch]     = p0;
        *(u32x4*)&lA[buf][rl * TSTR + ch + 8] = p1;
#ifndef HAVE_ASYNC_LDS
        *(u32x4*)&lB[buf][rl * TSTR + ch]     = fb0;
        *(u32x4*)&lB[buf][rl * TSTR + ch + 8] = fb1;
#endif
    };

    v8f acc[2][4];
    for (int a = 0; a < 2; a++)
        for (int b = 0; b < 4; b++)
            for (int r = 0; r < 8; r++) acc[a][b][r] = 0.0f;

    const int half = lane >> 4;
    const int lrow = lane & 15;

    const int nIt = Kpad >> 5;
    // prologue: tile 0 -> buffer 0
    fetchA(0);
    issueB(0, 0);
    storeTile(0);

    for (int it = 0; it < nIt; it++) {
        wait_async_all();       // own async B stores to LDS complete
        __syncthreads();        // all waves' tile data visible
        const bool more = (it + 1) < nIt;
        if (more) { fetchA((it + 1) << 5); issueB((it + 1) << 5, (it + 1) & 1); }

        const int buf = it & 1;
        // ---- fragments (ISA 16-bit A layout; B mirrored column-major) ----
        FragU afrag[2], bfrag[4];
        #pragma unroll
        for (int mi = 0; mi < 2; mi++) {
            const int r = wr * 32 + mi * 16 + lrow;
            afrag[mi].u[0] = *(const u32x4*)&lA[buf][r * TSTR + half * 8];
            afrag[mi].u[1] = *(const u32x4*)&lA[buf][r * TSTR + 16 + half * 8];
        }
        #pragma unroll
        for (int nj = 0; nj < 4; nj++) {
            const int c = wc * 64 + nj * 16 + lrow;
            bfrag[nj].u[0] = *(const u32x4*)&lB[buf][c * TSTR + half * 8];
            bfrag[nj].u[1] = *(const u32x4*)&lB[buf][c * TSTR + 16 + half * 8];
        }
        #pragma unroll
        for (int mi = 0; mi < 2; mi++)
            #pragma unroll
            for (int nj = 0; nj < 4; nj++)
                acc[mi][nj] = __builtin_amdgcn_wmma_f32_16x16x32_bf16(
                    false, afrag[mi].v, false, bfrag[nj].v,
                    (short)0, acc[mi][nj], false, false);

        if (more) storeTile((it + 1) & 1);
    }

    // ---- epilogue: bias + optional ReLU, f32 store ----
    #pragma unroll
    for (int mi = 0; mi < 2; mi++) {
        const long mb = m0 + wr * 32 + mi * 16 + 8 * half;
        #pragma unroll
        for (int nj = 0; nj < 4; nj++) {
            const int nb = n0 + wc * 64 + nj * 16 + lrow;
            const float bv = bias[nb];
            #pragma unroll
            for (int r = 0; r < 8; r++) {
                const long m = mb + r;
                if (m < (long)M) {
                    float v = acc[mi][nj][r] + bv;
                    if (relu) v = fmaxf(v, 0.0f);
                    out[m * (long)Nout + nb] = v;
                }
            }
        }
    }
}

// ---------------------------------------------------------------------------
// Helper kernels
// ---------------------------------------------------------------------------
__global__ void wprep_kernel(const float* W, unsigned short* wt, int K, int Nout, int Kpad) {
    int t = blockIdx.x * 256 + threadIdx.x;
    if (t >= Nout * Kpad) return;
    int n = t / Kpad, k = t % Kpad;
    wt[t] = (k < K) ? f2bf(W[(size_t)k * Nout + n]) : (unsigned short)0;
}

__global__ void fzero_kernel(float* p, long n) {
    long t = (long)blockIdx.x * blockDim.x + threadIdx.x;
    if (t < n) p[t] = 0.0f;
}
__global__ void izero_kernel(int* p, long n) {
    long t = (long)blockIdx.x * blockDim.x + threadIdx.x;
    if (t < n) p[t] = 0;
}

// BatchNorm (training mode, biased variance): per-column scale/shift
__global__ void bn_stats_kernel(const float* in, int R, int C,
                                const float* gamma, const float* beta,
                                float* scale, float* shift) {
    const int c = blockIdx.x;
    float s = 0.0f, q = 0.0f;
    for (int r = threadIdx.x; r < R; r += blockDim.x) {
        float v = in[(size_t)r * C + c];
        s += v; q += v * v;
    }
    __shared__ float ss[256], qq[256];
    ss[threadIdx.x] = s; qq[threadIdx.x] = q;
    __syncthreads();
    for (int st = 128; st > 0; st >>= 1) {
        if (threadIdx.x < st) { ss[threadIdx.x] += ss[threadIdx.x + st]; qq[threadIdx.x] += qq[threadIdx.x + st]; }
        __syncthreads();
    }
    if (threadIdx.x == 0) {
        float mean = ss[0] / (float)R;
        float var  = qq[0] / (float)R - mean * mean;
        float inv  = rsqrtf(var + 1e-5f);
        float sc   = gamma[c] * inv;
        scale[c] = sc;
        shift[c] = beta[c] - mean * sc;
    }
}

__global__ void bn_apply_kernel(const float* in, float* out, long total, int C,
                                const float* scale, const float* shift) {
    long t = (long)blockIdx.x * blockDim.x + threadIdx.x;
    if (t >= total) return;
    int c = (int)(t % C);
    out[t] = in[t] * scale[c] + shift[c];
}

__global__ void count_edges_kernel(const int* col, int* cnt, int E) {
    int t = blockIdx.x * blockDim.x + threadIdx.x;
    if (t < E) atomicAdd(&cnt[col[t]], 1);
}

// scatter-add m[E,256] into agg[col[e], 256]
__global__ void scatter_add_kernel(const float* m, const int* col, float* agg, long total) {
    long t = (long)blockIdx.x * blockDim.x + threadIdx.x;
    if (t >= total) return;
    int e = (int)(t >> 8);
    int f = (int)(t & 255);
    atomicAdd(&agg[(size_t)col[e] * 256 + f], m[t]);
}

__global__ void agg_div_kernel(float* agg, const int* cnt, long total) {
    long t = (long)blockIdx.x * blockDim.x + threadIdx.x;
    if (t >= total) return;
    agg[t] /= fmaxf((float)cnt[t >> 8], 1.0f);
}

// per-graph mean of x[N,128]; nodes of graph b are rows [b*64, b*64+64)
__global__ void graph_mean_kernel(const float* x, float* nm) {
    int b = blockIdx.x, f = threadIdx.x;
    float s = 0.0f;
    for (int i = 0; i < 64; i++) s += x[(size_t)(b * 64 + i) * 128 + f];
    nm[b * 128 + f] = s * (1.0f / 64.0f);
}

// concat [u | node_mean] into zero-padded stride-Ws buffer (Ws = ceil32(ud+128))
__global__ void build_gcat_kernel(const float* u, int ud, const float* nm,
                                  float* gc, int B, int Ws) {
    int t = blockIdx.x * blockDim.x + threadIdx.x;
    if (t >= B * Ws) return;
    int b = t / Ws, c = t % Ws;
    float v = 0.0f;
    if (c < ud) v = u[b * ud + c];
    else if (c < ud + 128) v = nm[b * 128 + (c - ud)];
    gc[t] = v;
}

// y[b] = e[sel[2b]] + e[sel[2b+1]]  (e width 128 after e6)
__global__ void select_sum_kernel(const float* e, const int* sel, float* y, int B) {
    int t = blockIdx.x * blockDim.x + threadIdx.x;
    if (t >= B * 128) return;
    int b = t >> 7, f = t & 127;
    y[t] = e[(size_t)sel[2 * b] * 128 + f] + e[(size_t)sel[2 * b + 1] * 128 + f];
}

// out[b] = dot(hy[b,:], W[:,0]) + bias[0]
__global__ void head_final_kernel(const float* hy, const float* W, const float* b, float* out) {
    int g = blockIdx.x, t = threadIdx.x;  // 128 threads
    __shared__ float red[128];
    red[t] = hy[(size_t)g * 128 + t] * W[t];
    __syncthreads();
    for (int s = 64; s > 0; s >>= 1) {
        if (t < s) red[t] += red[t + s];
        __syncthreads();
    }
    if (t == 0) out[g] = red[0] + b[0];
}

// ---------------------------------------------------------------------------
// Host side
// ---------------------------------------------------------------------------
struct Lin { const float* W; const float* b; int K; int Nout; unsigned short* wt; };

extern "C" void kernel_launch(void* const* d_in, const int* in_sizes, int n_in,
                              void* d_out, int out_size, void* d_ws, size_t ws_size,
                              hipStream_t stream) {
    (void)in_sizes; (void)n_in; (void)out_size; (void)ws_size;
    const int Bg = 256, Nn = 256 * 64, Ee = 256 * 512, NF = 32, EF = 16;

    // ---- parse d_in (setup_inputs dict insertion order, depth-first) ----
    int cur = 0;
    auto F = [&]() -> const float* { return (const float*)d_in[cur++]; };
    const float* bnN_g = F(); const float* bnN_b = F();
    const float* bnE_g = F(); const float* bnE_b = F();

    Lin eL[6][3], naL[5][3], nbL[5][2], gL[5][3], hdL[2];
    auto rd = [&](Lin* Ls, int n, const int* dims) {
        for (int i = 0; i < n; i++) {
            Ls[i].W = F(); Ls[i].b = F();
            Ls[i].K = dims[i]; Ls[i].Nout = dims[i + 1]; Ls[i].wt = nullptr;
        }
    };
    { int d[4] = {80, 128, 128, 512};  rd(eL[0], 3, d); }
    { int d[4] = {544, 256, 256, 256}; rd(naL[0], 3, d); }
    { int d[3] = {288, 256, 128};      rd(nbL[0], 2, d); }
    { int d[4] = {129, 256, 256, 128}; rd(gL[0], 3, d); }
    for (int l = 1; l < 5; l++) {
        int de[4] = {768, 128, 128, 512}; rd(eL[l], 3, de);
        int da[4] = {640, 256, 256, 256}; rd(naL[l], 3, da);
        int db[3] = {384, 256, 128};      rd(nbL[l], 2, db);
        int dg[4] = {256, 256, 256, 128}; rd(gL[l], 3, dg);
    }
    { int d[4] = {768, 128, 128, 128}; rd(eL[5], 3, d); }
    { int d[3] = {128, 128, 1};        rd(hdL, 2, d); }

    const float* xin = F();
    const float* ein = F();
    const float* gin = F();
    const int* eidx  = (const int*)d_in[cur++];
    const int* batch = (const int*)d_in[cur++]; (void)batch;
    const int* sel   = (const int*)d_in[cur++];
    const int* ebarr = (const int*)d_in[cur++]; (void)ebarr;
    const int* rowi = eidx;
    const int* coli = eidx + Ee;

    // ---- workspace carve ----
    char* wsp = (char*)d_ws;
    size_t off = 0;
    auto alloc = [&](size_t bytes) -> void* {
        void* p = wsp + off;
        off = (off + bytes + 255) & ~(size_t)255;
        return p;
    };
    float* x    = (float*)alloc((size_t)Nn * 128 * 4);
    float* e    = (float*)alloc((size_t)Ee * 512 * 4);
    float* tA   = (float*)alloc((size_t)Ee * 256 * 4);
    float* tB   = (float*)alloc((size_t)Ee * 256 * 4);
    float* agg  = (float*)alloc((size_t)Nn * 256 * 4);
    int*   cnt  = (int*)  alloc((size_t)Nn * 4);
    float* u    = (float*)alloc((size_t)Bg * 128 * 4);
    float* nm   = (float*)alloc((size_t)Bg * 128 * 4);
    float* gcat = (float*)alloc((size_t)Bg * 256 * 4);
    float* hgA  = (float*)alloc((size_t)Bg * 256 * 4);
    float* hgB  = (float*)alloc((size_t)Bg * 256 * 4);
    float* ybuf = (float*)alloc((size_t)Bg * 128 * 4);
    float* hy   = (float*)alloc((size_t)Bg * 128 * 4);
    float* bnNsc = (float*)alloc(NF * 4); float* bnNsh = (float*)alloc(NF * 4);
    float* bnEsc = (float*)alloc(EF * 4); float* bnEsh = (float*)alloc(EF * 4);

    // ---- weight prep: fp32 [K,Nout] -> bf16 transposed [Nout,Kpad] ----
    auto prep = [&](Lin& L) {
        int Kp = (L.K + 31) & ~31;
        L.wt = (unsigned short*)alloc((size_t)L.Nout * Kp * 2);
        int tot = L.Nout * Kp;
        wprep_kernel<<<(tot + 255) / 256, 256, 0, stream>>>(L.W, L.wt, L.K, L.Nout, Kp);
    };
    for (int l = 0; l < 6; l++) for (int i = 0; i < 3; i++) prep(eL[l][i]);
    for (int l = 0; l < 5; l++) {
        for (int i = 0; i < 3; i++) prep(naL[l][i]);
        for (int i = 0; i < 2; i++) prep(nbL[l][i]);
        for (int i = 0; i < 3; i++) prep(gL[l][i]);
    }
    prep(hdL[0]);

    auto gemm = [&](const float* a0, int w0, const int* i0,
                    const float* a1, int w1, const int* i1,
                    const float* a2, int w2, const int* i2,
                    const Lin& L, float* out, int M, int relu) {
        dim3 grid(L.Nout / 128, M / 128);
        gemm_bf16_seg<<<grid, 256, 0, stream>>>(a0, w0, i0, a1, w1, i1, a2, w2, i2,
                                                L.wt, L.b, out, M, w0 + w1 + w2,
                                                L.Nout, relu);
    };

    // ---- BatchNorm ----
    bn_stats_kernel<<<NF, 256, 0, stream>>>(xin, Nn, NF, bnN_g, bnN_b, bnNsc, bnNsh);
    bn_apply_kernel<<<((long)Nn * NF + 255) / 256, 256, 0, stream>>>(xin, x, (long)Nn * NF, NF, bnNsc, bnNsh);
    bn_stats_kernel<<<EF, 256, 0, stream>>>(ein, Ee, EF, bnE_g, bnE_b, bnEsc, bnEsh);
    bn_apply_kernel<<<((long)Ee * EF + 255) / 256, 256, 0, stream>>>(ein, e, (long)Ee * EF, EF, bnEsc, bnEsh);

    // ---- in-degree counts (col), fixed across layers ----
    izero_kernel<<<(Nn + 255) / 256, 256, 0, stream>>>(cnt, Nn);
    count_edges_kernel<<<(Ee + 255) / 256, 256, 0, stream>>>(coli, cnt, Ee);

    // ---- meta layers 1..5 ----
    for (int l = 0; l < 5; l++) {
        const int xd = (l == 0) ? 32 : 128;
        const int ed = (l == 0) ? 16 : 512;
        // EdgeModel: e = MLP([x[row], x[col], e])
        gemm(x, xd, rowi, x, xd, coli, e, ed, nullptr, eL[l][0], tA, Ee, 1);
        gemm(tA, 128, nullptr, nullptr, 0, nullptr, nullptr, 0, nullptr, eL[l][1], tB, Ee, 1);
        gemm(tB, 128, nullptr, nullptr, 0, nullptr, nullptr, 0, nullptr, eL[l][2], e, Ee, 0);
        // NodeModel part a: m = MLP([x[row], e])
        gemm(x, xd, rowi, e, 512, nullptr, nullptr, 0, nullptr, naL[l][0], tA, Ee, 1);
        gemm(tA, 256, nullptr, nullptr, 0, nullptr, nullptr, 0, nullptr, naL[l][1], tB, Ee, 1);
        gemm(tB, 256, nullptr, nullptr, 0, nullptr, nullptr, 0, nullptr, naL[l][2], tA, Ee, 0);
        // agg = scatter_mean(m, col)
        fzero_kernel<<<((long)Nn * 256 + 255) / 256, 256, 0, stream>>>(agg, (long)Nn * 256);
        scatter_add_kernel<<<((long)Ee * 256 + 255) / 256, 256, 0, stream>>>(tA, coli, agg, (long)Ee * 256);
        agg_div_kernel<<<((long)Nn * 256 + 255) / 256, 256, 0, stream>>>(agg, cnt, (long)Nn * 256);
        // NodeModel part b: x = MLP([x, agg])
        gemm(x, xd, nullptr, agg, 256, nullptr, nullptr, 0, nullptr, nbL[l][0], tB, Nn, 1);
        gemm(tB, 256, nullptr, nullptr, 0, nullptr, nullptr, 0, nullptr, nbL[l][1], x, Nn, 0);
        // GlobalModel: u = MLP([u, graph_mean(x)])
        graph_mean_kernel<<<Bg, 128, 0, stream>>>(x, nm);
        const float* up = (l == 0) ? gin : u;
        const int ud = (l == 0) ? 1 : 128;
        const int Ws = (ud + 128 + 31) & ~31;  // 160 or 256 (zero-padded concat)
        build_gcat_kernel<<<(Bg * Ws + 255) / 256, 256, 0, stream>>>(up, ud, nm, gcat, Bg, Ws);
        gemm(gcat, Ws, nullptr, nullptr, 0, nullptr, nullptr, 0, nullptr, gL[l][0], hgA, Bg, 1);
        gemm(hgA, 256, nullptr, nullptr, 0, nullptr, nullptr, 0, nullptr, gL[l][1], hgB, Bg, 1);
        gemm(hgB, 256, nullptr, nullptr, 0, nullptr, nullptr, 0, nullptr, gL[l][2], u, Bg, 0);
    }

    // ---- meta6: edge model only (output width 128) ----
    gemm(x, 128, rowi, x, 128, coli, e, 512, nullptr, eL[5][0], tA, Ee, 1);
    gemm(tA, 128, nullptr, nullptr, 0, nullptr, nullptr, 0, nullptr, eL[5][1], tB, Ee, 1);
    gemm(tB, 128, nullptr, nullptr, 0, nullptr, nullptr, 0, nullptr, eL[5][2], e, Ee, 0);

    // ---- select 2 edges/graph, sum, head MLP ----
    select_sum_kernel<<<(Bg * 128 + 255) / 256, 256, 0, stream>>>(e, sel, ybuf, Bg);
    gemm(ybuf, 128, nullptr, nullptr, 0, nullptr, nullptr, 0, nullptr, hdL[0], hy, Bg, 1);
    head_final_kernel<<<Bg, 128, 0, stream>>>(hy, hdL[1].W, hdL[1].b, (float*)d_out);
}
